// VideoRelation_59055800320098
// MI455X (gfx1250) — compile-verified, matrix operand order checked by reading, not run on hardware
//
#include <hip/hip_runtime.h>

#define FEAT 2048
#define QN   4096
#define KN   512
#define BSZ  2
#define NG   16
#define DG   128

typedef __attribute__((ext_vector_type(16))) _Float16 v16h;
typedef __attribute__((ext_vector_type(8)))  _Float16 v8h;
typedef __attribute__((ext_vector_type(8)))  float    v8f;

union Frag32 { v16h v; uint4 u[2]; };

__device__ __forceinline__ v8f wmma32(v16h a, v16h b, v8f c) {
  return __builtin_amdgcn_wmma_f32_16x16x32_f16(false, a, false, b, (short)0, c,
                                                false, false);
}

// CDNA5 async DMA: global -> LDS, no VGPR staging, tracked by ASYNCcnt.
__device__ __forceinline__ void async_copy16(unsigned lds, const _Float16* g) {
  asm volatile("global_load_async_to_lds_b128 %0, %1, off"
               :: "v"(lds), "v"(g) : "memory");
}
__device__ __forceinline__ void wait_async0() {
  asm volatile("s_wait_asynccnt 0x0" ::: "memory");
}

// ---------------- fp32 -> fp16 cast (8 elems / thread) ----------------
__global__ void cast_f32_f16(const float* __restrict__ s, _Float16* __restrict__ d,
                             long n) {
  long i = ((long)blockIdx.x * blockDim.x + threadIdx.x) * 8;
  if (i + 8 > n) return;
  float4 a = *(const float4*)(s + i);
  float4 b = *(const float4*)(s + i + 4);
  v8h h;
  h[0] = (_Float16)a.x; h[1] = (_Float16)a.y; h[2] = (_Float16)a.z; h[3] = (_Float16)a.w;
  h[4] = (_Float16)b.x; h[5] = (_Float16)b.y; h[6] = (_Float16)b.z; h[7] = (_Float16)b.w;
  *(v8h*)(d + i) = h;
}

// ---------------- tiled WMMA GEMM: C[m,n] = act(A[m,:] . Bw[n,:] + bias[n]) ----
// A: [M x FEAT] f16 row-major, Bw: [FEAT x FEAT] f16 row-major (weights, N-major)
// Double-buffered LDS, tiles filled by async global->LDS DMA.
__global__ __launch_bounds__(256)
void gemm_wmma(const _Float16* __restrict__ A, const _Float16* __restrict__ Bw,
               const float* __restrict__ bias,
               float* __restrict__ out32, _Float16* __restrict__ out16,
               _Float16* __restrict__ outT, int ldT, int relu) {
  __shared__ __align__(16) _Float16 sA[2][128 * 32];
  __shared__ __align__(16) _Float16 sB[2][128 * 32];
  const int tid  = threadIdx.x;
  const int lane = tid & 31;
  const int wid  = tid >> 5;
  const int wm   = wid >> 2;           // 0..1
  const int wn   = wid & 3;            // 0..3
  const long bm  = (long)blockIdx.x * 128;
  const long bn  = (long)blockIdx.y * 128;
  const int l16  = lane & 15;
  const int koff = (lane < 16) ? 0 : 8;

  // per-thread DMA slots: rows row0 and row0+64, 8-f16 chunk ch0 (16B)
  const int row0 = tid >> 2;           // 0..63
  const int ch0  = (tid & 3) << 3;     // 0,8,16,24

  const v8f vz = {0.f, 0.f, 0.f, 0.f, 0.f, 0.f, 0.f, 0.f};
  v8f acc[4][2];
#pragma unroll
  for (int i = 0; i < 4; ++i)
#pragma unroll
    for (int j = 0; j < 2; ++j) acc[i][j] = vz;

  auto load_tile = [&](int buf, int k0) {
    const _Float16* gA = &A[(bm + row0) * FEAT + k0 + ch0];
    const _Float16* gB = &Bw[(bn + row0) * FEAT + k0 + ch0];
    unsigned la = (unsigned)(uintptr_t)(&sA[buf][row0 * 32 + ch0]);
    unsigned lb = (unsigned)(uintptr_t)(&sB[buf][row0 * 32 + ch0]);
    async_copy16(la,               gA);
    async_copy16(la + 64 * 32 * 2, gA + (long)64 * FEAT);
    async_copy16(lb,               gB);
    async_copy16(lb + 64 * 32 * 2, gB + (long)64 * FEAT);
  };

  load_tile(0, 0);
  wait_async0();
  __syncthreads();

  for (int k0 = 0; k0 < FEAT; k0 += 32) {
    const int buf = (k0 >> 5) & 1;
    if (k0 + 32 < FEAT) load_tile(buf ^ 1, k0 + 32);   // DMA next tile

    Frag32 af[4], bf[2];
#pragma unroll
    for (int i = 0; i < 4; ++i) {
      int r = wm * 64 + i * 16 + l16;
      af[i].u[0] = *(const uint4*)(&sA[buf][r * 32 + koff]);
      af[i].u[1] = *(const uint4*)(&sA[buf][r * 32 + koff + 16]);
    }
#pragma unroll
    for (int j = 0; j < 2; ++j) {
      int r = wn * 32 + j * 16 + l16;
      bf[j].u[0] = *(const uint4*)(&sB[buf][r * 32 + koff]);
      bf[j].u[1] = *(const uint4*)(&sB[buf][r * 32 + koff + 16]);
    }
#pragma unroll
    for (int i = 0; i < 4; ++i)
#pragma unroll
      for (int j = 0; j < 2; ++j)
        acc[i][j] = wmma32(af[i].v, bf[j].v, acc[i][j]);

    // next tile landed; barrier also guards WAR on buf^1 for iteration k0+64
    wait_async0();
    __syncthreads();
  }

  const int moff = (lane < 16) ? 0 : 8;
#pragma unroll
  for (int i = 0; i < 4; ++i) {
#pragma unroll
    for (int j = 0; j < 2; ++j) {
      long n  = bn + wn * 32 + j * 16 + l16;
      long mb = bm + wm * 64 + i * 16 + moff;
      float bv = bias ? bias[n] : 0.f;
      float vals[8];
#pragma unroll
      for (int r = 0; r < 8; ++r) {
        float x = acc[i][j][r] + bv;
        if (relu) x = fmaxf(x, 0.f);
        vals[r] = x;
      }
      if (out32) {
#pragma unroll
        for (int r = 0; r < 8; ++r) out32[(mb + r) * FEAT + n] = vals[r];
      }
      if (out16) {
#pragma unroll
        for (int r = 0; r < 8; ++r) out16[(mb + r) * FEAT + n] = (_Float16)vals[r];
      }
      if (outT) {
        v8h hh;
#pragma unroll
        for (int r = 0; r < 8; ++r) hh[r] = (_Float16)vals[r];
        *(v8h*)(&outT[n * (long)ldT + mb]) = hh;   // mb multiple of 8 -> 16B aligned
      }
    }
  }
}

// ---------------- fused flash attention + residual + b_v --------------
// Qm: [BSZ*QN x FEAT] f16 (head h at cols h*DG..), Km: [BSZ*KN x FEAT] f16,
// Vt: [FEAT x BSZ*KN] f16 (transposed V' = sup1 @ Wv^T), out: f32 (+=).
// One wave handles 16 q rows of one (b,h). Computes S^T = K.Q^T so the
// softmaxed P converts to an A-fragment with per-lane packs only.
__global__ __launch_bounds__(128)
void attn_wmma(const _Float16* __restrict__ Qm, const _Float16* __restrict__ Km,
               const _Float16* __restrict__ Vt, const float* __restrict__ bv,
               float* __restrict__ out) {
  const int lane  = threadIdx.x & 31;
  const int wid   = threadIdx.x >> 5;
  const int h     = blockIdx.y;
  const int b     = blockIdx.z;
  const int qtile = blockIdx.x * 4 + wid;          // 0..255
  const long qbase = (long)b * QN + (long)qtile * 16;
  const int l16  = lane & 15;
  const int koff = (lane < 16) ? 0 : 8;
  const int moff = (lane < 16) ? 0 : 8;

  // Q fragments, used as B (Q^T): lane = q row, regs = dims
  Frag32 qf[4];
  {
    const _Float16* qrow = Qm + (qbase + l16) * FEAT + h * DG;
#pragma unroll
    for (int d = 0; d < 4; ++d) {
      qf[d].u[0] = *(const uint4*)(qrow + d * 32 + koff);
      qf[d].u[1] = *(const uint4*)(qrow + d * 32 + koff + 16);
    }
  }

  const v8f vz = {0.f, 0.f, 0.f, 0.f, 0.f, 0.f, 0.f, 0.f};
  v8f o[8];
#pragma unroll
  for (int t = 0; t < 8; ++t) o[t] = vz;
  float m_st = -3.0e38f, l_st = 0.f;
  const float scale = 0.08838834764831845f;        // 1/sqrt(128)

  for (int kc = 0; kc < KN; kc += 32) {
    // S^T tiles: s0 = keys kc..kc+15, s1 = keys kc+16..kc+31 (lane = q)
    v8f s0 = vz, s1 = vz;
    {
      const _Float16* krow = Km + ((long)b * KN + kc + l16) * FEAT + h * DG;
#pragma unroll
      for (int d = 0; d < 4; ++d) {
        Frag32 kf;
        kf.u[0] = *(const uint4*)(krow + d * 32 + koff);
        kf.u[1] = *(const uint4*)(krow + d * 32 + koff + 16);
        s0 = wmma32(kf.v, qf[d].v, s0);
      }
      krow += (long)16 * FEAT;
#pragma unroll
      for (int d = 0; d < 4; ++d) {
        Frag32 kf;
        kf.u[0] = *(const uint4*)(krow + d * 32 + koff);
        kf.u[1] = *(const uint4*)(krow + d * 32 + koff + 16);
        s1 = wmma32(kf.v, qf[d].v, s1);
      }
    }
    // online softmax over this 32-key chunk (row = q = lane%16)
    float mloc = -3.0e38f;
#pragma unroll
    for (int r = 0; r < 8; ++r) {
      s0[r] *= scale; s1[r] *= scale;
      mloc = fmaxf(mloc, fmaxf(s0[r], s1[r]));
    }
    mloc = fmaxf(mloc, __shfl_xor(mloc, 16));
    float mnew  = fmaxf(m_st, mloc);
    float alpha = __expf(m_st - mnew);
    float rs = 0.f;
#pragma unroll
    for (int r = 0; r < 8; ++r) {
      s0[r] = __expf(s0[r] - mnew); rs += s0[r];
      s1[r] = __expf(s1[r] - mnew); rs += s1[r];
    }
    rs += __shfl_xor(rs, 16);
    l_st = l_st * alpha + rs;
    m_st = mnew;
    // pack P as A-fragment (lane = q, regs = 32 keys) — per-lane only
    Frag32 pf;
#pragma unroll
    for (int r = 0; r < 8; ++r) {
      pf.v[r]     = (_Float16)s0[r];
      pf.v[8 + r] = (_Float16)s1[r];
    }
    // broadcast alpha to O layout (vgpr r <-> q = r + moff)
    float aq[8];
#pragma unroll
    for (int r = 0; r < 8; ++r) aq[r] = __shfl(alpha, r + moff);
#pragma unroll
    for (int t = 0; t < 8; ++t) {
#pragma unroll
      for (int r = 0; r < 8; ++r) o[t][r] *= aq[r];
      const _Float16* vrow =
          Vt + ((long)(h * DG + t * 16 + l16)) * (BSZ * KN) + (long)b * KN + kc;
      Frag32 vf;
      vf.u[0] = *(const uint4*)(vrow + koff);
      vf.u[1] = *(const uint4*)(vrow + koff + 16);
      o[t] = wmma32(pf.v, vf.v, o[t]);
    }
  }

  float linv[8];
#pragma unroll
  for (int r = 0; r < 8; ++r) linv[r] = 1.0f / __shfl(l_st, r + moff);
#pragma unroll
  for (int t = 0; t < 8; ++t) {
    int col = h * DG + t * 16 + l16;
    float bb = bv[col];
#pragma unroll
    for (int r = 0; r < 8; ++r) {
      long idx = (qbase + moff + r) * FEAT + col;
      out[idx] += o[t][r] * linv[r] + bb;
    }
  }
}

extern "C" void kernel_launch(void* const* d_in, const int* in_sizes, int n_in,
                              void* d_out, int out_size, void* d_ws, size_t ws_size,
                              hipStream_t stream) {
  (void)in_sizes; (void)n_in; (void)out_size; (void)ws_size;
  const float* ref_feat = (const float*)d_in[0];
  const float* sup_feat = (const float*)d_in[1];
  const float* W_fc = (const float*)d_in[2];
  const float* b_fc = (const float*)d_in[3];
  const float* W_q  = (const float*)d_in[4];
  const float* b_q  = (const float*)d_in[5];
  const float* W_k  = (const float*)d_in[6];
  const float* b_k  = (const float*)d_in[7];
  const float* W_v  = (const float*)d_in[8];
  const float* b_v  = (const float*)d_in[9];
  float* out = (float*)d_out;

  size_t off = 0;
  auto carve = [&](size_t bytes) -> void* {
    void* p = (char*)d_ws + off;
    off += (bytes + 255) & ~(size_t)255;
    return p;
  };
  _Float16* ref16   = (_Float16*)carve((size_t)BSZ * QN * FEAT * 2);
  _Float16* sup16   = (_Float16*)carve((size_t)BSZ * KN * FEAT * 2);
  _Float16* wfc16   = (_Float16*)carve((size_t)FEAT * FEAT * 2);
  _Float16* wq16    = (_Float16*)carve((size_t)FEAT * FEAT * 2);
  _Float16* wk16    = (_Float16*)carve((size_t)FEAT * FEAT * 2);
  _Float16* wv16    = (_Float16*)carve((size_t)FEAT * FEAT * 2);
  _Float16* ref1_16 = (_Float16*)carve((size_t)BSZ * QN * FEAT * 2);
  _Float16* sup1_16 = (_Float16*)carve((size_t)BSZ * KN * FEAT * 2);
  _Float16* q16     = (_Float16*)carve((size_t)BSZ * QN * FEAT * 2);
  _Float16* k16     = (_Float16*)carve((size_t)BSZ * KN * FEAT * 2);
  _Float16* vT16    = (_Float16*)carve((size_t)FEAT * BSZ * KN * 2);

  auto cast = [&](const float* s, _Float16* d, long n) {
    cast_f32_f16<<<dim3((unsigned)(n / 8 / 256)), dim3(256), 0, stream>>>(s, d, n);
  };
  cast(ref_feat, ref16, (long)BSZ * QN * FEAT);
  cast(sup_feat, sup16, (long)BSZ * KN * FEAT);
  cast(W_fc, wfc16, (long)FEAT * FEAT);
  cast(W_q, wq16, (long)FEAT * FEAT);
  cast(W_k, wk16, (long)FEAT * FEAT);
  cast(W_v, wv16, (long)FEAT * FEAT);

  dim3 blk(256);
  // ref1 = relu(ref @ Wfc^T + b_fc): f32 -> d_out (residual), f16 -> ref1_16
  gemm_wmma<<<dim3(BSZ * QN / 128, FEAT / 128), blk, 0, stream>>>(
      ref16, wfc16, b_fc, out, ref1_16, nullptr, 0, 1);
  // sup1 = relu(sup @ Wfc^T + b_fc)
  gemm_wmma<<<dim3(BSZ * KN / 128, FEAT / 128), blk, 0, stream>>>(
      sup16, wfc16, b_fc, nullptr, sup1_16, nullptr, 0, 1);
  // q = ref1 @ Wq^T + b_q
  gemm_wmma<<<dim3(BSZ * QN / 128, FEAT / 128), blk, 0, stream>>>(
      ref1_16, wq16, b_q, nullptr, q16, nullptr, 0, 0);
  // k = sup1 @ Wk^T + b_k
  gemm_wmma<<<dim3(BSZ * KN / 128, FEAT / 128), blk, 0, stream>>>(
      sup1_16, wk16, b_k, nullptr, k16, nullptr, 0, 0);
  // V' = sup1 @ Wv^T, stored transposed [FEAT x BSZ*KN]
  gemm_wmma<<<dim3(BSZ * KN / 128, FEAT / 128), blk, 0, stream>>>(
      sup1_16, wv16, nullptr, nullptr, nullptr, vT16, BSZ * KN, 0);
  // attention + grouped-conv (pre-folded into V') + b_v + residual
  attn_wmma<<<dim3(QN / 64, NG, BSZ), dim3(128), 0, stream>>>(
      q16, k16, vT16, b_v, out);
}
